// LSTM_20160576487812
// MI455X (gfx1250) — compile-verified, matrix operand order checked by reading, not run on hardware
//
#include <hip/hip_runtime.h>
#include <math.h>

typedef __attribute__((ext_vector_type(16))) _Float16 v16h;
typedef __attribute__((ext_vector_type(8)))  _Float16 v8h;
typedef __attribute__((ext_vector_type(4)))  _Float16 v4h;
typedef __attribute__((ext_vector_type(8)))  float    v8f;

#define kH   128        // hidden size
#define kAS  264        // xh LDS row stride in halfs (128 x + 128 h + 8 pad)
#define kFS  132        // fo LDS row stride in floats (128 + 4 pad)

// ---- branch-free hardware transcendentals (critical path of the scan) -------
#if __has_builtin(__builtin_amdgcn_exp2f)
#define EXP2F(x) __builtin_amdgcn_exp2f(x)
#else
#define EXP2F(x) exp2f(x)
#endif
#if __has_builtin(__builtin_amdgcn_rcpf)
#define RCPF(x) __builtin_amdgcn_rcpf(x)
#else
#define RCPF(x) (1.0f / (x))
#endif

__device__ __forceinline__ float fast_tanh(float x) {
#if __has_builtin(__builtin_amdgcn_tanhf)
    return __builtin_amdgcn_tanhf(x);          // V_TANH_F32 (gfx1250 trans op)
#else
    // tanh(x) = sign(x) * (1 - e)/(1 + e), e = 2^(-2*log2(e)*|x|); no branches
    const float ax = __builtin_fabsf(x);
    const float e  = EXP2F(-2.8853900817779268f * ax);
    const float r  = (1.0f - e) * RCPF(1.0f + e);
    return __builtin_copysignf(r, x);
#endif
}

__device__ __forceinline__ float fast_sigmoid(float x) {
    // 1 / (1 + 2^(-x*log2(e))): saturates to 0/1 at extremes, no branches
    return RCPF(1.0f + EXP2F(-1.4426950408889634f * x));
}

__global__ __launch_bounds__(512)
void lstm_scan_kernel(const float* __restrict__ x,
                      const float* __restrict__ h0,
                      const float* __restrict__ c0,
                      const float* __restrict__ Wih,
                      const float* __restrict__ Whh,
                      const float* __restrict__ bih,
                      const float* __restrict__ bhh,
                      float* __restrict__ out,
                      int T)
{
    // A-operand staging: [16 batch rows][cols 0..127 = x_t (f16), 128..255 = h (f16)]
    __shared__ _Float16 xh[16 * kAS];
    // sigmoid(f) / sigmoid(o) exchange buffers, f32: [2][16][kFS]
    __shared__ float fo[2 * 16 * kFS];

    const int tid  = threadIdx.x;
    const int lane = tid & 31;
    const int w    = tid >> 5;              // wave id 0..15
    const int b0   = blockIdx.x * 16;       // batch slice base row
    const int lr   = lane & 15;             // 0..15
    const int hi   = (lane >> 4) & 1;       // lane half
    const int hcol = (w & 7) * 16 + lr;     // hidden column owned by wave pair

    // ---------------- Preload B fragments (weights, f16) into registers -------
    // Wave w owns N-tiles {w, w+16}: tiles 0-7=i, 8-15=f, 16-23=g, 24-31=o.
    // Combined K=256: k<128 -> W_ih, k>=128 -> W_hh (both stored [4H][H] row-major).
    v16h Bf[2][8];
    float bias[2];
    #pragma unroll
    for (int nt = 0; nt < 2; ++nt) {
        const int tile = w + 16 * nt;
        const int n = tile * 16 + lr;                 // gate column 0..511
        bias[nt] = bih[n] + bhh[n];
        #pragma unroll
        for (int kt = 0; kt < 8; ++kt) {
            v16h b;
            #pragma unroll
            for (int j = 0; j < 16; ++j) {
                const int krel = hi * 8 + (j & 7) + ((j >> 3) << 4);
                const int k = kt * 32 + krel;
                const float wv = (k < kH) ? Wih[(size_t)n * kH + k]
                                          : Whh[(size_t)n * kH + (k - kH)];
                b[j] = (_Float16)wv;
            }
            Bf[nt][kt] = b;
        }
    }

    // ---------------- Init h (LDS, f16) and x_0 (LDS, f16) --------------------
    {
        const int r = w;                 // 512 threads: row = wave, col = lane*4
        const int c = lane * 4;
        const float4 hv = *reinterpret_cast<const float4*>(&h0[(size_t)(b0 + r) * kH + c]);
        v4h hp; hp[0] = (_Float16)hv.x; hp[1] = (_Float16)hv.y;
                hp[2] = (_Float16)hv.z; hp[3] = (_Float16)hv.w;
        *reinterpret_cast<v4h*>(&xh[r * kAS + kH + c]) = hp;

        const float4 xv = *reinterpret_cast<const float4*>(
            &x[((size_t)(b0 + r) * T + 0) * kH + c]);
        v4h xp; xp[0] = (_Float16)xv.x; xp[1] = (_Float16)xv.y;
                xp[2] = (_Float16)xv.z; xp[3] = (_Float16)xv.w;
        *reinterpret_cast<v4h*>(&xh[r * kAS + c]) = xp;
    }

    // ---------------- Init c (registers, waves 0..7, D-matrix layout) ---------
    v8f creg = {};
    if (w < 8) {
        #pragma unroll
        for (int r = 0; r < 8; ++r) {
            const int row = hi ? (r + 8) : r;
            creg[r] = c0[(size_t)(b0 + row) * kH + hcol];
        }
    }

    // Incrementally-advanced prefetch pointer (x_{t+1}, clamped at the end).
    const float* xp = &x[((size_t)(b0 + w) * T + ((T > 1) ? 1 : 0)) * kH + lane * 4];
    // Per-lane output base for r=0 rows; advance by kH each step.
    float* outBase = &out[(size_t)(b0 + hi * 8) * T * kH + hcol];
    size_t tOff = 0;

    // A-fragment LDS base for this lane (bytes handled by indexing in halfs).
    const int aBase = lr * kAS + hi * 8;

    __syncthreads();

    // ---------------- Scan -----------------------------------------------------
    #pragma unroll 1
    for (int t = 0; t < T; ++t) {
        // Prefetch x_{t+1} (recurrence-independent): overlap with GEMM+barrier.
        const float4 xnext = *reinterpret_cast<const float4*>(xp);
        xp = (t + 2 < T) ? (xp + kH) : xp;      // uniform scalar select

        // GEMM: g[16,32] = [x_t | h][16,256] @ Wc^T tiles, +bias, f32 accum.
        v8f acc0, acc1;
        #pragma unroll
        for (int r = 0; r < 8; ++r) { acc0[r] = bias[0]; acc1[r] = bias[1]; }

        #pragma unroll
        for (int kt = 0; kt < 8; ++kt) {
            const int off = aBase + kt * 32;
            const v8h a0 = *reinterpret_cast<const v8h*>(&xh[off]);
            const v8h a1 = *reinterpret_cast<const v8h*>(&xh[off + 16]);
            v16h a;
            #pragma unroll
            for (int j = 0; j < 8; ++j) { a[j] = a0[j]; a[j + 8] = a1[j]; }

            acc0 = __builtin_amdgcn_wmma_f32_16x16x32_f16(
                false, a, false, Bf[0][kt], (short)0, acc0, false, false);
            acc1 = __builtin_amdgcn_wmma_f32_16x16x32_f16(
                false, a, false, Bf[1][kt], (short)0, acc1, false, false);
        }

#if __has_builtin(__builtin_amdgcn_sched_group_barrier)
        // Force a software-pipelined schedule for this block:
        //   4 DS reads ; 6 x (2 WMMA ; 2 DS reads) ; 4 WMMA
        // (16 ds_load_b128 + 16 v_wmma total). DS-read mask=0x100, WMMA=0x008.
        // Loads for stage kt+1 issue before stage kt's WMMAs -> overlapping
        // live ranges -> distinct VGPRs -> partial s_wait_dscnt waits.
        __builtin_amdgcn_sched_group_barrier(0x100, 4, 0);
        #pragma unroll
        for (int i = 0; i < 6; ++i) {
            __builtin_amdgcn_sched_group_barrier(0x008, 2, 0);
            __builtin_amdgcn_sched_group_barrier(0x100, 2, 0);
        }
        __builtin_amdgcn_sched_group_barrier(0x008, 4, 0);
#endif

        // Waves 8..15: publish sigmoid(f), sigmoid(o) for partner wave.
        if (w >= 8) {
            #pragma unroll
            for (int r = 0; r < 8; ++r) {
                const int row = hi ? (r + 8) : r;
                fo[row * kFS + hcol]            = fast_sigmoid(acc0[r]);  // f gate
                fo[16 * kFS + row * kFS + hcol] = fast_sigmoid(acc1[r]);  // o gate
            }
        }
        __syncthreads();

        // Waves 0..7: c/h update, write h (LDS f16) and output (global f32).
        if (w < 8) {
            #pragma unroll
            for (int r = 0; r < 8; ++r) {
                const int row = hi ? (r + 8) : r;
                const float iv = fast_sigmoid(acc0[r]);   // i gate
                const float gv = fast_tanh(acc1[r]);      // g gate
                const float fv = fo[row * kFS + hcol];
                const float ov = fo[16 * kFS + row * kFS + hcol];
                const float cn = fv * creg[r] + iv * gv;
                creg[r] = cn;
                const float hn = ov * fast_tanh(cn);
                xh[row * kAS + kH + hcol] = (_Float16)hn;
                outBase[(size_t)r * T * kH + tOff] = hn;
            }
        }

        // All threads: commit prefetched x_{t+1} into LDS x-region.
        {
            v4h xq; xq[0] = (_Float16)xnext.x; xq[1] = (_Float16)xnext.y;
                    xq[2] = (_Float16)xnext.z; xq[3] = (_Float16)xnext.w;
            *reinterpret_cast<v4h*>(&xh[w * kAS + lane * 4]) = xq;
        }
        tOff += kH;
        __syncthreads();
    }
}

extern "C" void kernel_launch(void* const* d_in, const int* in_sizes, int n_in,
                              void* d_out, int out_size, void* d_ws, size_t ws_size,
                              hipStream_t stream) {
    const float* x   = (const float*)d_in[0];
    const float* h0  = (const float*)d_in[1];
    const float* c0  = (const float*)d_in[2];
    const float* Wih = (const float*)d_in[3];
    const float* Whh = (const float*)d_in[4];
    const float* bih = (const float*)d_in[5];
    const float* bhh = (const float*)d_in[6];
    float* out = (float*)d_out;

    const int B = in_sizes[1] / kH;                  // h0 is [1,B,H]
    const int T = in_sizes[0] / (B * kH);            // x is [B,T,H]
    const int nblocks = B / 16;                      // one 16-row batch slice per WG

    lstm_scan_kernel<<<nblocks, 512, 0, stream>>>(x, h0, c0, Wih, Whh,
                                                  bih, bhh, out, T);
}
